// attn_head_21165598834965
// MI455X (gfx1250) — matrix operand order repaired and checked by reading, not compiled, not run on hardware
//
#include <hip/hip_runtime.h>

typedef __bf16 bf16;
typedef __attribute__((ext_vector_type(16))) __bf16 v16bf;
typedef __attribute__((ext_vector_type(8)))  __bf16 v8bf;
typedef __attribute__((ext_vector_type(8)))  float  v8f;
typedef __attribute__((ext_vector_type(4)))  float  v4f;

#define N_NODES 8192
#define F_IN    128
#define H_DIM   64
#define KSPLIT  8
#define QBLOCKS (N_NODES / 64)          // 128 query blocks of 64 rows

static __device__ __forceinline__ v16bf concat8(v8bf lo, v8bf hi) {
  return __builtin_shufflevector(lo, hi, 0,1,2,3,4,5,6,7,8,9,10,11,12,13,14,15);
}

// -------------------------------------------------------------------------
// Kernel 0: zero the f32 accumulators (graph-replay safe re-init)
// -------------------------------------------------------------------------
__global__ void __launch_bounds__(256) zero_kernel(float* __restrict__ o_ws,
                                                   float* __restrict__ den_ws) {
  const int idx = blockIdx.x * 256 + threadIdx.x;
  if (idx < N_NODES * H_DIM) o_ws[idx] = 0.0f;
  if (idx < N_NODES)         den_ws[idx] = 0.0f;
}

// -------------------------------------------------------------------------
// Kernel 1: Q = (seq@W1)/8, K = seq@W2, Vt = (seq@W1)^T   (all bf16)
// 8 waves / block: waves 0-3 -> W1 (q + vt, n-tiles 0..3), waves 4-7 -> W2.
// -------------------------------------------------------------------------
__global__ void __launch_bounds__(256) proj_kernel(
    const float* __restrict__ seq, const float* __restrict__ W1,
    const float* __restrict__ W2, bf16* __restrict__ qb,
    bf16* __restrict__ kb, bf16* __restrict__ vtb) {
  const int lane = threadIdx.x & 31;
  const int wave = threadIdx.x >> 5;
  const int mat  = wave >> 2;            // 0: W1 (q + vt), 1: W2 (k)
  const int h0   = (wave & 3) * 16;
  const int nlo  = lane & 15;
  const bool hi  = lane >= 16;

  const float* W = mat ? W2 : W1;

  v16bf bw[4];                            // B-frags of W (128x64), K chunks of 32
#pragma unroll
  for (int c = 0; c < 4; ++c) {
    const int kbase = 32 * c + (hi ? 16 : 0);
    v16bf b;
#pragma unroll
    for (int j = 0; j < 16; ++j)
      b[j] = (bf16)W[(kbase + j) * H_DIM + h0 + nlo];
    bw[c] = b;
  }

  for (int t = blockIdx.x; t < N_NODES / 16; t += gridDim.x) {
    const int row0 = t * 16;
    const float* srow = seq + (size_t)(row0 + nlo) * F_IN;
    v8f acc = {};
#pragma unroll
    for (int c = 0; c < 4; ++c) {
      const int koff = 32 * c + (hi ? 16 : 0);
      v16bf a;
#pragma unroll
      for (int j = 0; j < 8; ++j) {
        a[j]     = (bf16)srow[koff + j];
        a[8 + j] = (bf16)srow[koff + 8 + j];
      }
      acc = __builtin_amdgcn_wmma_f32_16x16x32_bf16(false, a, false, bw[c],
                                                    (short)0, acc, false, false);
    }
    const int rbase = row0 + (hi ? 8 : 0);
    if (mat == 0) {
      v8bf vv;
#pragma unroll
      for (int i = 0; i < 8; ++i) {
        qb[(rbase + i) * H_DIM + h0 + nlo] = (bf16)(acc[i] * 0.125f); // fold 1/sqrt(64)
        vv[i] = (bf16)acc[i];
      }
      *(v8bf*)(vtb + (size_t)(h0 + nlo) * N_NODES + rbase) = vv;      // V^T
    } else {
#pragma unroll
      for (int i = 0; i < 8; ++i)
        kb[(rbase + i) * H_DIM + h0 + nlo] = (bf16)acc[i];
    }
  }
}

// -------------------------------------------------------------------------
// Kernel 2: fused exp-attention over a key slice; register-only P transpose.
// S^T = K·Q^T  =>  C-frag (lane = query, elems = keys) == A-frag of P for P·V.
// Bias stream uses non-temporal loads (TH=NT) so the 256 MB single-pass
// stream does not evict the L2-resident Q/K/V^T working set.
// Partial num/den accumulated to f32 workspace via atomics (linear combine).
// -------------------------------------------------------------------------
__global__ void __launch_bounds__(128) attn_kernel(
    const float* __restrict__ bias, const bf16* __restrict__ qb,
    const bf16* __restrict__ kb, const bf16* __restrict__ vtb,
    float* __restrict__ o_ws, float* __restrict__ den_ws) {
  const int lane = threadIdx.x & 31;
  const int wave = threadIdx.x >> 5;
  const int nlo  = lane & 15;
  const bool hi  = lane >= 16;

  const int qblk = blockIdx.x & (QBLOCKS - 1);
  const int kgrp = blockIdx.x / QBLOCKS;
  const int q0   = qblk * 64 + wave * 16;
  const int kbeg = kgrp * (N_NODES / KSPLIT);
  const int kend = kbeg + (N_NODES / KSPLIT);

  // Q^T B-fragments (loop-invariant): B[h, m] = Qs[q0+m][h]
  v16bf bq[2];
  {
    const bf16* qrow = qb + (size_t)(q0 + nlo) * H_DIM;
#pragma unroll
    for (int c = 0; c < 2; ++c)
      bq[c] = *(const v16bf*)(qrow + 32 * c + (hi ? 16 : 0));
  }

  v8f o[4] = {{}, {}, {}, {}};
  float dsum = 0.0f;
  const float* brow = bias + (size_t)(q0 + nlo) * N_NODES;

  for (int kt = kbeg; kt < kend; kt += 32) {
    v8f p[2];
#pragma unroll
    for (int nt = 0; nt < 2; ++nt) {
      // A-frag of K tile: rows = keys, K-dim = h (two chunks of 32)
      const bf16* krow = kb + (size_t)(kt + nt * 16 + nlo) * H_DIM;
      v8f s = {};
#pragma unroll
      for (int c = 0; c < 2; ++c) {
        const int koff = 32 * c + (hi ? 16 : 0);
        v16bf ak = concat8(*(const v8bf*)(krow + koff),
                           *(const v8bf*)(krow + koff + 8));
        s = __builtin_amdgcn_wmma_f32_16x16x32_bf16(false, ak, false, bq[c],
                                                    (short)0, s, false, false);
      }
      // bias: 8 consecutive floats per lane -> two non-temporal b128 loads
      const float* bp = brow + kt + nt * 16 + (hi ? 8 : 0);
      const v4f b0 = __builtin_nontemporal_load((const v4f*)bp);
      const v4f b1 = __builtin_nontemporal_load((const v4f*)(bp + 4));
      v8f bb = __builtin_shufflevector(b0, b1, 0, 1, 2, 3, 4, 5, 6, 7);
      v8f pv;
#pragma unroll
      for (int i = 0; i < 8; ++i) {
        pv[i] = __expf(s[i]) * bb[i];
        dsum += pv[i];
      }
      p[nt] = pv;
    }

    // P A-fragment built purely in registers (no LDS transpose needed)
    v16bf pa;
#pragma unroll
    for (int j = 0; j < 8; ++j) {
      pa[j]     = (bf16)p[0][j];
      pa[8 + j] = (bf16)p[1][j];
    }

    // O += P·V  (B-frags from V^T: two 16B chunks per lane, L2-resident)
    const bf16* vt = vtb + (size_t)nlo * N_NODES + kt + (hi ? 8 : 0);
#pragma unroll
    for (int nt2 = 0; nt2 < 4; ++nt2) {
      v16bf bv = concat8(*(const v8bf*)(vt + (size_t)(nt2 * 16) * N_NODES),
                         *(const v8bf*)(vt + (size_t)(nt2 * 16) * N_NODES + 16));
      o[nt2] = __builtin_amdgcn_wmma_f32_16x16x32_bf16(false, pa, false, bv,
                                                       (short)0, o[nt2], false, false);
    }
  }

  // ---- combine partials: num (O C-frag: lane = h col, rows = queries) ----
#pragma unroll
  for (int nt2 = 0; nt2 < 4; ++nt2) {
#pragma unroll
    for (int i = 0; i < 8; ++i)
      atomicAdd(&o_ws[(size_t)(q0 + i + (hi ? 8 : 0)) * H_DIM + nt2 * 16 + nlo],
                o[nt2][i]);
  }
  // den: each lane covers a disjoint key subset of query q0+nlo
  atomicAdd(&den_ws[q0 + nlo], dsum);
}

// -------------------------------------------------------------------------
// Kernel 3: out = elu(num / (den + 1e-19))
// -------------------------------------------------------------------------
__global__ void __launch_bounds__(256) finish_kernel(
    const float* __restrict__ o_ws, const float* __restrict__ den_ws,
    float* __restrict__ out) {
  const int idx = blockIdx.x * 256 + threadIdx.x;
  if (idx >= N_NODES * H_DIM) return;
  float v = o_ws[idx] / (den_ws[idx >> 6] + 1e-19f);
  out[idx] = v > 0.0f ? v : (__expf(v) - 1.0f);
}

extern "C" void kernel_launch(void* const* d_in, const int* in_sizes, int n_in,
                              void* d_out, int out_size, void* d_ws, size_t ws_size,
                              hipStream_t stream) {
  const float* seq  = (const float*)d_in[0];
  const float* bias = (const float*)d_in[1];
  const float* W1   = (const float*)d_in[2];
  const float* W2   = (const float*)d_in[3];
  float* out = (float*)d_out;

  bf16* qb  = (bf16*)d_ws;                          // 8192x64 bf16 (1 MB)
  bf16* kb  = qb + (size_t)N_NODES * H_DIM;         // 8192x64 bf16 (1 MB)
  bf16* vtb = kb + (size_t)N_NODES * H_DIM;         // 64x8192 bf16 (1 MB)
  float* o_ws   = (float*)(vtb + (size_t)N_NODES * H_DIM); // 8192x64 f32 (2 MB)
  float* den_ws = o_ws + (size_t)N_NODES * H_DIM;          // 8192 f32

  zero_kernel<<<(N_NODES * H_DIM + 255) / 256, 256, 0, stream>>>(o_ws, den_ws);
  proj_kernel<<<64, 256, 0, stream>>>(seq, W1, W2, qb, kb, vtb);
  attn_kernel<<<QBLOCKS * KSPLIT, 128, 0, stream>>>(bias, qb, kb, vtb, o_ws, den_ws);
  finish_kernel<<<(N_NODES * H_DIM + 255) / 256, 256, 0, stream>>>(o_ws, den_ws, out);
}
